// FourierTokenMixing_24111946400487
// MI455X (gfx1250) — matrix operand order repaired and checked by reading, not compiled, not run on hardware
//
#include <hip/hip_runtime.h>
#include <hip/hip_bf16.h>
#include <math.h>

// ---------------------------------------------------------------------------
// Types for CDNA5 WMMA (gfx1250, wave32)
// ---------------------------------------------------------------------------
typedef __bf16 bf16_t;
typedef __bf16  bf16x16 __attribute__((ext_vector_type(16)));
typedef float   f32x8   __attribute__((ext_vector_type(8)));

#define NPOS 8192   // L (sequence length / DCT size)
#define NCH  256    // C (channels)
#define NB   8      // batch

// split float into bf16 hi + bf16 lo (error term)
__device__ __forceinline__ void bsplit(float v, bf16_t& h, bf16_t& l) {
    h = (bf16_t)v;
    l = (bf16_t)(v - (float)h);
}

// load a 32-byte (16 x bf16) WMMA fragment from 16B-aligned LDS
__device__ __forceinline__ bf16x16 ldsFrag(const bf16_t* p) {
    union { uint4 q[2]; bf16x16 v; } u;
    u.q[0] = *reinterpret_cast<const uint4*>(p);
    u.q[1] = *reinterpret_cast<const uint4*>(p + 8);
    return u.v;
}

// ---------------------------------------------------------------------------
// CDNA5 async global->LDS DMA (16B per lane), tracked by ASYNCcnt.
// VDST = VGPR with LDS byte offset; VADDR = 64-bit global address.
// ---------------------------------------------------------------------------
__device__ __forceinline__ void asyncCopy16(bf16_t* lds_dst, const bf16_t* gsrc) {
    unsigned ldsOff = (unsigned)(unsigned long long)
        (__attribute__((address_space(3))) bf16_t*)lds_dst;
    asm volatile("global_load_async_to_lds_b128 %0, %1, off"
                 :: "v"(ldsOff), "v"((unsigned long long)gsrc)
                 : "memory");
}
__device__ __forceinline__ void waitAsync0() {
    asm volatile("s_wait_asynccnt 0x0" ::: "memory");
}
__device__ __forceinline__ void waitAsync6() {
    asm volatile("s_wait_asynccnt 0x6" ::: "memory");
}

// ---------------------------------------------------------------------------
// Kernel 1: LayerNorm over C, transpose to [b][c][l], bf16 hi/lo split
// ---------------------------------------------------------------------------
__global__ __launch_bounds__(256) void layernorm_tr_kernel(
    const float* __restrict__ x, const float* __restrict__ lnw,
    const float* __restrict__ lnb,
    bf16_t* __restrict__ xth, bf16_t* __restrict__ xtl) {
    __shared__ __attribute__((aligned(16))) bf16_t Th[NCH][32];
    __shared__ __attribute__((aligned(16))) bf16_t Tl[NCH][32];

    const int tid  = threadIdx.x;
    const int wave = tid >> 5, lane = tid & 31;
    const int b = blockIdx.y;
    const int l0 = blockIdx.x * 32;

    float g[8], be[8];
#pragma unroll
    for (int e = 0; e < 8; ++e) {
        g[e]  = lnw[lane * 8 + e];
        be[e] = lnb[lane * 8 + e];
    }

    for (int iter = 0; iter < 4; ++iter) {
        const int l = l0 + wave * 4 + iter;
        const float* row = x + ((long)b * NPOS + l) * NCH + lane * 8;
        float4 v0 = *reinterpret_cast<const float4*>(row);
        float4 v1 = *reinterpret_cast<const float4*>(row + 4);
        float e[8] = {v0.x, v0.y, v0.z, v0.w, v1.x, v1.y, v1.z, v1.w};
        float s = 0.f, s2 = 0.f;
#pragma unroll
        for (int i = 0; i < 8; ++i) { s += e[i]; s2 += e[i] * e[i]; }
#pragma unroll
        for (int off = 16; off > 0; off >>= 1) {
            s  += __shfl_xor(s,  off, 32);
            s2 += __shfl_xor(s2, off, 32);
        }
        const float mean = s * (1.0f / NCH);
        const float var  = s2 * (1.0f / NCH) - mean * mean;
        const float rstd = rsqrtf(var + 1e-5f);
        const int lrel = l - l0;
#pragma unroll
        for (int i = 0; i < 8; ++i) {
            const float v = (e[i] - mean) * rstd * g[i] + be[i];
            bf16_t h, lo;
            bsplit(v, h, lo);
            Th[lane * 8 + i][lrel] = h;
            Tl[lane * 8 + i][lrel] = lo;
        }
    }
    __syncthreads();
    for (int iter = 0; iter < 4; ++iter) {
        const int id = iter * 256 + tid;
        const int c = id >> 2, ch = id & 3;
        const long dst = ((long)b * NCH + c) * NPOS + l0 + ch * 8;
        *reinterpret_cast<uint4*>(xth + dst) = *reinterpret_cast<uint4*>(&Th[c][ch * 8]);
        *reinterpret_cast<uint4*>(xtl + dst) = *reinterpret_cast<uint4*>(&Tl[c][ch * 8]);
    }
}

// ---------------------------------------------------------------------------
// Kernel 2: DCT-II(ortho) basis D[k][n] and transpose, split hi/lo.
// Exact angle reduction: k*(2n+1) mod 4N (power of two) -> small __cosf args.
// ---------------------------------------------------------------------------
__device__ __forceinline__ float dct_entry(int k, int n) {
    const unsigned prod = (unsigned)k * (unsigned)(2 * n + 1);
    const unsigned q = prod & (4u * NPOS - 1u);
    const float ang = (float)q * (float)(M_PI / (2.0 * NPOS));
    const float coef = (k == 0) ? 0.011048543456039806f  // 1/sqrt(8192)
                                : 0.015625f;             // sqrt(2/8192)
    return coef * __cosf(ang);
}

__global__ __launch_bounds__(256) void gen_basis_kernel(
    bf16_t* __restrict__ Dh, bf16_t* __restrict__ Dl,
    bf16_t* __restrict__ DTh, bf16_t* __restrict__ DTl) {
    const int c = blockIdx.x * 256 + threadIdx.x;
    const int r = blockIdx.y;
    const long o = (long)r * NPOS + c;
    bf16_t h, l;
    bsplit(dct_entry(r, c), h, l);
    Dh[o] = h; Dl[o] = l;
    bsplit(dct_entry(c, r), h, l);
    DTh[o] = h; DTl[o] = l;
}

// ---------------------------------------------------------------------------
// Kernel 3: merge the 4 conv kernels into one centered 11-tap kernel [t][o][i]
// ---------------------------------------------------------------------------
__global__ __launch_bounds__(256) void pack_weights_kernel(
    const float* __restrict__ w3,  const float* __restrict__ b3,
    const float* __restrict__ w5,  const float* __restrict__ b5,
    const float* __restrict__ w7,  const float* __restrict__ b7,
    const float* __restrict__ w11, const float* __restrict__ b11,
    bf16_t* __restrict__ Wh, bf16_t* __restrict__ Wl,
    float* __restrict__ bsum) {
    const int idx = blockIdx.x * 256 + threadIdx.x;
    if (idx >= 11 * NCH * NCH) return;
    const int t = idx / (NCH * NCH);
    const int rem = idx - t * NCH * NCH;
    const int o = rem >> 8, i = rem & 255;
    const long oi = (long)o * NCH + i;
    float v = w11[oi * 11 + t];
    const int t7 = t - 2, t5 = t - 3, t3 = t - 4;
    if (t7 >= 0 && t7 < 7) v += w7[oi * 7 + t7];
    if (t5 >= 0 && t5 < 5) v += w5[oi * 5 + t5];
    if (t3 >= 0 && t3 < 3) v += w3[oi * 3 + t3];
    bf16_t h, l;
    bsplit(v, h, l);
    Wh[idx] = h; Wl[idx] = l;
    if (idx < NCH) bsum[idx] = b3[idx] + b5[idx] + b7[idx] + b11[idx];
}

// ---------------------------------------------------------------------------
// Split-precision bf16 WMMA GEMM with async-DMA double-buffered LDS.
//   C[M,N] = A[M,K] * B[K,N];  A row-major hi/lo planes, B^T [N][K] hi/lo.
//   out = AhBh + AhBl + AlBh (3 WMMAs / tile / k-step)
// WG = 256 threads = 8 waves; wave tile 32x32; WG tile 64(M) x 128(N).
// ---------------------------------------------------------------------------
#define LDT 40  // LDS row stride (elements); 80B keeps 16B alignment

template <bool STORE_F32>
__global__ __launch_bounds__(256) void gemm_split_kernel(
    const bf16_t* __restrict__ Ahi, const bf16_t* __restrict__ Alo,
    const bf16_t* __restrict__ Bthi, const bf16_t* __restrict__ Btlo,
    long bBatchStride,
    float* __restrict__ outF,
    bf16_t* __restrict__ outHi, bf16_t* __restrict__ outLo,
    long oBatchStride, int K, int ldo) {
    __shared__ __attribute__((aligned(16))) bf16_t As[2][2][64][LDT];
    __shared__ __attribute__((aligned(16))) bf16_t Bs[2][2][128][LDT];

    const int tid = threadIdx.x;
    const int b = blockIdx.z;
    const int mBase = blockIdx.y * 64;
    const int nBase = blockIdx.x * 128;
    const bf16_t* Bh = Bthi + (long)b * bBatchStride;
    const bf16_t* Bl = Btlo + (long)b * bBatchStride;

    const int wave = tid >> 5, lane = tid & 31;
    const int waveM = wave >> 2, waveN = wave & 3;
    const int lm = lane & 15, kh = lane >> 4;

    f32x8 acc[2][2];
#pragma unroll
    for (int i = 0; i < 2; ++i)
#pragma unroll
        for (int j = 0; j < 2; ++j)
#pragma unroll
            for (int e = 0; e < 8; ++e) acc[i][j][e] = 0.0f;

    const int ar = tid >> 2, ac = tid & 3;
    const int apos = ac;  // chunk order [0,1,2,3]
    const int br = tid >> 1, bhalf = tid & 1;

    // issue one k-step tile (6 async DMA instructions per wave)
    auto issueTiles = [&](int kk, int buf) {
        const long asrc = (long)(mBase + ar) * K + kk + ac * 8;
        asyncCopy16(&As[buf][0][ar][apos * 8], Ahi + asrc);
        asyncCopy16(&As[buf][1][ar][apos * 8], Alo + asrc);
        const long bsrc = (long)(nBase + br) * K + kk + bhalf * 16;
        asyncCopy16(&Bs[buf][0][br][bhalf * 16],     Bh + bsrc);
        asyncCopy16(&Bs[buf][0][br][bhalf * 16 + 8], Bh + bsrc + 8);
        asyncCopy16(&Bs[buf][1][br][bhalf * 16],     Bl + bsrc);
        asyncCopy16(&Bs[buf][1][br][bhalf * 16 + 8], Bl + bsrc + 8);
    };

    const int nsteps = K >> 5;
    issueTiles(0, 0);
    for (int i = 0; i < nsteps; ++i) {
        const int cur = i & 1;
        if (i + 1 < nsteps) {
            issueTiles((i + 1) << 5, cur ^ 1);
            waitAsync6();   // 6 just issued -> current tile's 6 are complete
        } else {
            waitAsync0();
        }
        __syncthreads();    // all waves' current tile visible

        bf16x16 ah[2], al[2], bhf[2], blf[2];
#pragma unroll
        for (int tm = 0; tm < 2; ++tm) {
            ah[tm] = ldsFrag(&As[cur][0][waveM * 32 + tm * 16 + lm][kh * 16]);
            al[tm] = ldsFrag(&As[cur][1][waveM * 32 + tm * 16 + lm][kh * 16]);
        }
#pragma unroll
        for (int tn = 0; tn < 2; ++tn) {
            bhf[tn] = ldsFrag(&Bs[cur][0][waveN * 32 + tn * 16 + lm][kh * 16]);
            blf[tn] = ldsFrag(&Bs[cur][1][waveN * 32 + tn * 16 + lm][kh * 16]);
        }
#pragma unroll
        for (int tm = 0; tm < 2; ++tm)
#pragma unroll
            for (int tn = 0; tn < 2; ++tn) {
                acc[tm][tn] = __builtin_amdgcn_wmma_f32_16x16x32_bf16(
                    false, ah[tm], false, bhf[tn], (short)0, acc[tm][tn], false, false);
                acc[tm][tn] = __builtin_amdgcn_wmma_f32_16x16x32_bf16(
                    false, ah[tm], false, blf[tn], (short)0, acc[tm][tn], false, false);
                acc[tm][tn] = __builtin_amdgcn_wmma_f32_16x16x32_bf16(
                    false, al[tm], false, bhf[tn], (short)0, acc[tm][tn], false, false);
            }
        __syncthreads();    // readers done before this buffer is DMA'd again
    }

    const int colLane = lane & 15, rowOff = (lane >> 4) * 8;
#pragma unroll
    for (int tm = 0; tm < 2; ++tm)
#pragma unroll
        for (int tn = 0; tn < 2; ++tn) {
            const int col = nBase + waveN * 32 + tn * 16 + colLane;
#pragma unroll
            for (int j = 0; j < 8; ++j) {
                const int row = mBase + waveM * 32 + tm * 16 + rowOff + j;
                const long o = (long)b * oBatchStride + (long)row * ldo + col;
                const float v = acc[tm][tn][j];
                if (STORE_F32) {
                    outF[o] = v;
                } else {
                    bf16_t h, l;
                    bsplit(v, h, l);
                    outHi[o] = h; outLo[o] = l;
                }
            }
        }
}

// ---------------------------------------------------------------------------
// Conv-as-GEMM: y[b][o][lf] = sum_t sum_i W[t][o][i] * xf[b][lf+t-5][i] + bias
// Async DMA with clamped rows; OOB slots patched to zero with ds_store so the
// per-wave ASYNCcnt stays uniform (counted waits remain valid).
// ---------------------------------------------------------------------------
__global__ __launch_bounds__(256) void conv_gemm_kernel(
    const bf16_t* __restrict__ Whi, const bf16_t* __restrict__ Wlo,
    const bf16_t* __restrict__ Xhi, const bf16_t* __restrict__ Xlo,
    const float* __restrict__ bsum,
    bf16_t* __restrict__ Yhi, bf16_t* __restrict__ Ylo) {
    __shared__ __attribute__((aligned(16))) bf16_t As[2][2][64][LDT];
    __shared__ __attribute__((aligned(16))) bf16_t Bs[2][2][128][LDT];

    const int tid = threadIdx.x;
    const int b = blockIdx.z;
    const int mBase = blockIdx.y * 64;   // output channel o
    const int nBase = blockIdx.x * 128;  // frequency position lf
    const long xbase = (long)b * NPOS * NCH;

    const int wave = tid >> 5, lane = tid & 31;
    const int waveM = wave >> 2, waveN = wave & 3;
    const int lm = lane & 15, kh = lane >> 4;

    f32x8 acc[2][2];
#pragma unroll
    for (int i = 0; i < 2; ++i)
#pragma unroll
        for (int j = 0; j < 2; ++j)
#pragma unroll
            for (int e = 0; e < 8; ++e) acc[i][j][e] = 0.0f;

    const int ar = tid >> 2, ac = tid & 3;
    const int apos = ac;
    const int br = tid >> 1, bhalf = tid & 1;
    const uint4 zero4 = make_uint4(0u, 0u, 0u, 0u);

    auto issueTiles = [&](int s, int buf) {
        const int t = s >> 3;
        const int kk = (s & 7) * 32;
        const long asrc = (long)t * NCH * NCH + (long)(mBase + ar) * NCH + kk + ac * 8;
        asyncCopy16(&As[buf][0][ar][apos * 8], Whi + asrc);
        asyncCopy16(&As[buf][1][ar][apos * 8], Wlo + asrc);
        int srcRow = nBase + br + t - 5;
        srcRow = srcRow < 0 ? 0 : (srcRow > NPOS - 1 ? NPOS - 1 : srcRow);
        const long bsrc = xbase + (long)srcRow * NCH + kk + bhalf * 16;
        asyncCopy16(&Bs[buf][0][br][bhalf * 16],     Xhi + bsrc);
        asyncCopy16(&Bs[buf][0][br][bhalf * 16 + 8], Xhi + bsrc + 8);
        asyncCopy16(&Bs[buf][1][br][bhalf * 16],     Xlo + bsrc);
        asyncCopy16(&Bs[buf][1][br][bhalf * 16 + 8], Xlo + bsrc + 8);
    };
    auto patchZeros = [&](int s, int buf) {
        const int t = s >> 3;
        const int srcRow = nBase + br + t - 5;
        if (srcRow < 0 || srcRow >= NPOS) {
            *reinterpret_cast<uint4*>(&Bs[buf][0][br][bhalf * 16])     = zero4;
            *reinterpret_cast<uint4*>(&Bs[buf][0][br][bhalf * 16 + 8]) = zero4;
            *reinterpret_cast<uint4*>(&Bs[buf][1][br][bhalf * 16])     = zero4;
            *reinterpret_cast<uint4*>(&Bs[buf][1][br][bhalf * 16 + 8]) = zero4;
        }
    };

    const int nsteps = 88;  // 11 taps x 8 k-steps
    issueTiles(0, 0);
    for (int s = 0; s < nsteps; ++s) {
        const int cur = s & 1;
        if (s + 1 < nsteps) {
            issueTiles(s + 1, cur ^ 1);
            waitAsync6();
        } else {
            waitAsync0();
        }
        patchZeros(s, cur);  // after DMA complete, before barrier
        __syncthreads();

        bf16x16 ah[2], al[2], bhf[2], blf[2];
#pragma unroll
        for (int tm = 0; tm < 2; ++tm) {
            ah[tm] = ldsFrag(&As[cur][0][waveM * 32 + tm * 16 + lm][kh * 16]);
            al[tm] = ldsFrag(&As[cur][1][waveM * 32 + tm * 16 + lm][kh * 16]);
        }
#pragma unroll
        for (int tn = 0; tn < 2; ++tn) {
            bhf[tn] = ldsFrag(&Bs[cur][0][waveN * 32 + tn * 16 + lm][kh * 16]);
            blf[tn] = ldsFrag(&Bs[cur][1][waveN * 32 + tn * 16 + lm][kh * 16]);
        }
#pragma unroll
        for (int tm = 0; tm < 2; ++tm)
#pragma unroll
            for (int tn = 0; tn < 2; ++tn) {
                acc[tm][tn] = __builtin_amdgcn_wmma_f32_16x16x32_bf16(
                    false, ah[tm], false, bhf[tn], (short)0, acc[tm][tn], false, false);
                acc[tm][tn] = __builtin_amdgcn_wmma_f32_16x16x32_bf16(
                    false, ah[tm], false, blf[tn], (short)0, acc[tm][tn], false, false);
                acc[tm][tn] = __builtin_amdgcn_wmma_f32_16x16x32_bf16(
                    false, al[tm], false, bhf[tn], (short)0, acc[tm][tn], false, false);
            }
        __syncthreads();
    }

    const int colLane = lane & 15, rowOff = (lane >> 4) * 8;
#pragma unroll
    for (int tm = 0; tm < 2; ++tm)
#pragma unroll
        for (int tn = 0; tn < 2; ++tn) {
            const int col = nBase + waveN * 32 + tn * 16 + colLane;
#pragma unroll
            for (int j = 0; j < 8; ++j) {
                const int row = mBase + waveM * 32 + tm * 16 + rowOff + j;  // o
                const float v = acc[tm][tn][j] + bsum[row];
                bf16_t h, l;
                bsplit(v, h, l);
                const long o = (long)b * NCH * NPOS + (long)row * NPOS + col;
                Yhi[o] = h; Ylo[o] = l;
            }
        }
}

// ---------------------------------------------------------------------------
// Host-side orchestration
// ---------------------------------------------------------------------------
extern "C" void kernel_launch(void* const* d_in, const int* in_sizes, int n_in,
                              void* d_out, int out_size, void* d_ws, size_t ws_size,
                              hipStream_t stream) {
    const float* x    = (const float*)d_in[0];
    const float* ln_w = (const float*)d_in[1];
    const float* ln_b = (const float*)d_in[2];
    const float* w3   = (const float*)d_in[3];
    const float* b3   = (const float*)d_in[4];
    const float* w5   = (const float*)d_in[5];
    const float* b5   = (const float*)d_in[6];
    const float* w7   = (const float*)d_in[7];
    const float* b7   = (const float*)d_in[8];
    const float* w11  = (const float*)d_in[9];
    const float* b11  = (const float*)d_in[10];
    float* out = (float*)d_out;

    char* ws = (char*)d_ws;
    const size_t SZ_D = (size_t)NPOS * NPOS * sizeof(bf16_t);       // 128 MB
    const size_t SZ_P = (size_t)NB * NPOS * NCH * sizeof(bf16_t);   // 32 MB
    const size_t SZ_W = (size_t)11 * NCH * NCH * sizeof(bf16_t);
    size_t off = 0;
    bf16_t* Dhi  = (bf16_t*)(ws + off); off += SZ_D;
    bf16_t* Dlo  = (bf16_t*)(ws + off); off += SZ_D;
    bf16_t* DThi = (bf16_t*)(ws + off); off += SZ_D;
    bf16_t* DTlo = (bf16_t*)(ws + off); off += SZ_D;
    bf16_t* xth  = (bf16_t*)(ws + off); off += SZ_P;
    bf16_t* xtl  = (bf16_t*)(ws + off); off += SZ_P;
    bf16_t* xfh  = (bf16_t*)(ws + off); off += SZ_P;
    bf16_t* xfl  = (bf16_t*)(ws + off); off += SZ_P;
    bf16_t* yh   = (bf16_t*)(ws + off); off += SZ_P;
    bf16_t* yl   = (bf16_t*)(ws + off); off += SZ_P;
    bf16_t* Wh   = (bf16_t*)(ws + off); off += SZ_W;
    bf16_t* Wl   = (bf16_t*)(ws + off); off += SZ_W;
    float*  bs   = (float*)(ws + off);  off += NCH * sizeof(float);

    gen_basis_kernel<<<dim3(NPOS / 256, NPOS), 256, 0, stream>>>(Dhi, Dlo, DThi, DTlo);
    pack_weights_kernel<<<(11 * NCH * NCH + 255) / 256, 256, 0, stream>>>(
        w3, b3, w5, b5, w7, b7, w11, b11, Wh, Wl, bs);
    layernorm_tr_kernel<<<dim3(NPOS / 32, NB), 256, 0, stream>>>(x, ln_w, ln_b, xth, xtl);
    // DCT:  Xf[b][k][c] = D[k][:] . xt[b][c][:]
    gemm_split_kernel<false><<<dim3(NCH / 128, NPOS / 64, NB), 256, 0, stream>>>(
        Dhi, Dlo, xth, xtl, (long)NCH * NPOS,
        nullptr, xfh, xfl, (long)NPOS * NCH, NPOS, NCH);
    // merged 11-tap conv along frequency axis
    conv_gemm_kernel<<<dim3(NPOS / 128, NCH / 64, NB), 256, 0, stream>>>(
        Wh, Wl, xfh, xfl, bs, yh, yl);
    // IDCT: out[b][lt][c] = DT[lt][:] . y[b][c][:]   (fp32 final output)
    gemm_split_kernel<true><<<dim3(NCH / 128, NPOS / 64, NB), 256, 0, stream>>>(
        DThi, DTlo, yh, yl, (long)NCH * NPOS,
        out, nullptr, nullptr, (long)NPOS * NCH, NPOS, NCH);
}
